// SDTPair_59682865545617
// MI455X (gfx1250) — matrix operand order repaired and checked by reading, not compiled, not run on hardware
//
#include <hip/hip_runtime.h>
#include <cstdint>
#include <cstddef>

// ---------------- problem constants ----------------
#define BB   4
#define TT   1024
#define DMODEL 1024
#define HH   16
#define DHH  64
#define FF   4096
#define FPP  1024
#define KSEL 256

typedef __bf16 bf16_t;
typedef __attribute__((ext_vector_type(16))) __bf16 v16bf;
typedef __attribute__((ext_vector_type(8)))  float  v8f;

__device__ __forceinline__ uint32_t bfbits(float f) {
  union { float f; uint32_t u; } c; c.f = f;
  uint32_t u = c.u;
  return (u + 0x7FFFu + ((u >> 16) & 1u)) >> 16;   // round-to-nearest-even
}
__device__ __forceinline__ bf16_t f2bf(float f) {
  unsigned short s = (unsigned short)bfbits(f);
  return __builtin_bit_cast(bf16_t, s);
}
__device__ __forceinline__ uint32_t pack2(float lo, float hi) {
  return (bfbits(lo) & 0xFFFFu) | (bfbits(hi) << 16);
}
__device__ __forceinline__ float fast_rcp(float x) { return __builtin_amdgcn_rcpf(x); }
__device__ __forceinline__ float fast_rsq(float x) { return __builtin_amdgcn_rsqf(x); }

// Wave-relative LDS byte offset of a __shared__ object (generic LDS addr low 32 bits).
__device__ __forceinline__ uint32_t lds_off(const void* p) {
  return (uint32_t)(uintptr_t)p;
}
__device__ __forceinline__ void async_wait() {
  asm volatile("s_wait_asynccnt 0x0" ::: "memory");
}

// Build a v16bf fragment from two 16B chunks (elements in memory order).
struct Frag32 { uint4 lo, hi; };
__device__ __forceinline__ v16bf make_frag(uint4 lo, uint4 hi) {
  Frag32 f; f.lo = lo; f.hi = hi;
  return __builtin_bit_cast(v16bf, f);
}
__device__ __forceinline__ v8f wmma_bf16(v16bf a, v16bf b, v8f c) {
  return __builtin_amdgcn_wmma_f32_16x16x32_bf16(false, a, false, b, (short)0, c, false, false);
}

// ---------------- weight packing: fp32 [K][N] -> pair-packed tiles ----------------
// Wp dword index o = (s*N + n)*16 + kp  (s = k/32, kp = k-pair within slab)
//   Wp[o] = { bf16(W[32s+2kp][n]), bf16(W[32s+2kp+1][n]) }
// -> the LDS B-tile for any (k0, n0) block is one contiguous run of N_tile*16 dwords.
__global__ void pack_weight_kernel(const float* __restrict__ W, uint32_t* __restrict__ Wp,
                                   int nshift, size_t total) {
  size_t o  = (size_t)blockIdx.x * blockDim.x + threadIdx.x;
  size_t st = (size_t)gridDim.x * blockDim.x;
  size_t N = (size_t)1 << nshift;
  for (; o < total; o += st) {
    int    kp = (int)(o & 15);
    size_t n  = (o >> 4) & (N - 1);
    size_t s  = o >> (4 + nshift);
    size_t row = s * 32 + 2 * kp;
    Wp[o] = pack2(W[row * N + n], W[(row + 1) * N + n]);
  }
}

// ---------------- generic GEMM: C[M,N] = A[M,K]*Bw[K,N] (+res), bf16 in, f32 out ----------
// Block tile 32x256, 8 waves; wave owns a 32x32 patch (2m x 2n subtiles).
// Double-buffered LDS tiles filled by async global->LDS copies (ASYNCcnt).
template<bool HAS_RES>
__global__ __launch_bounds__(256) void gemm_bf16_kernel(
    const bf16_t* __restrict__ A, const uint32_t* __restrict__ Bp,
    const float* __restrict__ res, float* __restrict__ C,
    int M, int N, int Kd)
{
  __shared__ bf16_t   As[2][32 * 32];
  __shared__ uint32_t Bt[2][256 * 16];
  int m0 = blockIdx.x * 32;
  int n0 = blockIdx.y * 256;
  int tid = threadIdx.x, wave = tid >> 5, lane = tid & 31;
  v8f acc[4] = {{}, {}, {}, {}};     // [mi*2+nj]

  auto stage = [&](int buf, int k0) {
    // B tile: 16KB contiguous; 4 x b128 per thread
    uint32_t bdst = lds_off(&Bt[buf][0]) + (uint32_t)tid * 64u;
    unsigned long long bsrc =
        (unsigned long long)(uintptr_t)(Bp + (((size_t)(k0 >> 5) * N + n0) << 4) + (size_t)tid * 16);
    asm volatile("global_load_async_to_lds_b128 %0, %1, off"           :: "v"(bdst), "v"(bsrc) : "memory");
    asm volatile("global_load_async_to_lds_b128 %0, %1, off offset:16" :: "v"(bdst), "v"(bsrc) : "memory");
    asm volatile("global_load_async_to_lds_b128 %0, %1, off offset:32" :: "v"(bdst), "v"(bsrc) : "memory");
    asm volatile("global_load_async_to_lds_b128 %0, %1, off offset:48" :: "v"(bdst), "v"(bsrc) : "memory");
    // A tile: 32 rows x 64B; one b64 per thread
    int r = tid >> 3, cch = tid & 7;
    uint32_t adst = lds_off(&As[buf][0]) + (uint32_t)(r * 64 + cch * 8);
    unsigned long long asrc =
        (unsigned long long)(uintptr_t)(&A[(size_t)(m0 + r) * Kd + k0 + cch * 4]);
    asm volatile("global_load_async_to_lds_b64 %0, %1, off" :: "v"(adst), "v"(asrc) : "memory");
  };

  stage(0, 0);
  async_wait();
  __syncthreads();

  for (int k0 = 0; k0 < Kd; k0 += 32) {
    int cur = (k0 >> 5) & 1;
    if (k0 + 32 < Kd) stage(cur ^ 1, k0 + 32);
    {
      int ml  = lane & 15;
      int kb  = (lane >> 4) << 3;
      int kp0 = (lane >> 4) << 2;
      const bf16_t* row0 = &As[cur][ml * 32];
      const bf16_t* row1 = &As[cur][(16 + ml) * 32];
      v16bf a0 = make_frag(*(const uint4*)(row0 + kb), *(const uint4*)(row0 + kb + 16));
      v16bf a1 = make_frag(*(const uint4*)(row1 + kb), *(const uint4*)(row1 + kb + 16));
      int nt0 = wave * 2;
#pragma unroll
      for (int nj = 0; nj < 2; ++nj) {
        const uint4* bp = (const uint4*)(&Bt[cur][((nt0 + nj) * 16 + ml) * 16 + kp0]);
        v16bf b = make_frag(bp[0], bp[2]);
        acc[0 * 2 + nj] = wmma_bf16(a0, b, acc[0 * 2 + nj]);
        acc[1 * 2 + nj] = wmma_bf16(a1, b, acc[1 * 2 + nj]);
      }
    }
    async_wait();
    __syncthreads();
  }

  int mbase = m0 + ((lane >> 4) * 8);
#pragma unroll
  for (int mi = 0; mi < 2; ++mi) {
#pragma unroll
    for (int nj = 0; nj < 2; ++nj) {
      int n = n0 + (wave * 2 + nj) * 16 + (lane & 15);
      v8f a = acc[mi * 2 + nj];
#pragma unroll
      for (int v = 0; v < 8; ++v) {
        int m = mbase + mi * 16 + v;
        float val = a[v];
        if (HAS_RES) val += res[(size_t)m * N + n];
        C[(size_t)m * N + n] = val;
      }
    }
  }
}

// ---------------- dual-weight SwiGLU GEMM: out = bf16(silu(A*G) * (A*U)) ----------------
// Block tile 32x128; wave w owns n-subtile w for both m-subtiles and both weights.
__global__ __launch_bounds__(256) void gemm_swiglu_kernel(
    const bf16_t* __restrict__ A, const uint32_t* __restrict__ Gp, const uint32_t* __restrict__ Up,
    bf16_t* __restrict__ Out, int M, int N, int Kd)
{
  __shared__ bf16_t   As[2][32 * 32];
  __shared__ uint32_t Bgt[2][128 * 16];
  __shared__ uint32_t But[2][128 * 16];
  int m0 = blockIdx.x * 32;
  int n0 = blockIdx.y * 128;
  int tid = threadIdx.x, wave = tid >> 5, lane = tid & 31;
  v8f ag0 = {}, ag1 = {}, au0 = {}, au1 = {};

  auto stage = [&](int buf, int k0) {
    size_t wsrc = (((size_t)(k0 >> 5) * N + n0) << 4) + (size_t)tid * 8;
    uint32_t gdst = lds_off(&Bgt[buf][0]) + (uint32_t)tid * 32u;
    unsigned long long gsrc = (unsigned long long)(uintptr_t)(Gp + wsrc);
    asm volatile("global_load_async_to_lds_b128 %0, %1, off"           :: "v"(gdst), "v"(gsrc) : "memory");
    asm volatile("global_load_async_to_lds_b128 %0, %1, off offset:16" :: "v"(gdst), "v"(gsrc) : "memory");
    uint32_t udst = lds_off(&But[buf][0]) + (uint32_t)tid * 32u;
    unsigned long long usrc = (unsigned long long)(uintptr_t)(Up + wsrc);
    asm volatile("global_load_async_to_lds_b128 %0, %1, off"           :: "v"(udst), "v"(usrc) : "memory");
    asm volatile("global_load_async_to_lds_b128 %0, %1, off offset:16" :: "v"(udst), "v"(usrc) : "memory");
    int r = tid >> 3, cch = tid & 7;
    uint32_t adst = lds_off(&As[buf][0]) + (uint32_t)(r * 64 + cch * 8);
    unsigned long long asrc =
        (unsigned long long)(uintptr_t)(&A[(size_t)(m0 + r) * Kd + k0 + cch * 4]);
    asm volatile("global_load_async_to_lds_b64 %0, %1, off" :: "v"(adst), "v"(asrc) : "memory");
  };

  stage(0, 0);
  async_wait();
  __syncthreads();

  for (int k0 = 0; k0 < Kd; k0 += 32) {
    int cur = (k0 >> 5) & 1;
    if (k0 + 32 < Kd) stage(cur ^ 1, k0 + 32);
    {
      int ml  = lane & 15;
      int kb  = (lane >> 4) << 3;
      int kp0 = (lane >> 4) << 2;
      const bf16_t* row0 = &As[cur][ml * 32];
      const bf16_t* row1 = &As[cur][(16 + ml) * 32];
      v16bf a0 = make_frag(*(const uint4*)(row0 + kb), *(const uint4*)(row0 + kb + 16));
      v16bf a1 = make_frag(*(const uint4*)(row1 + kb), *(const uint4*)(row1 + kb + 16));
      const uint4* gp = (const uint4*)(&Bgt[cur][(wave * 16 + ml) * 16 + kp0]);
      const uint4* up = (const uint4*)(&But[cur][(wave * 16 + ml) * 16 + kp0]);
      v16bf bg = make_frag(gp[0], gp[2]);
      v16bf bu = make_frag(up[0], up[2]);
      ag0 = wmma_bf16(a0, bg, ag0);
      ag1 = wmma_bf16(a1, bg, ag1);
      au0 = wmma_bf16(a0, bu, au0);
      au1 = wmma_bf16(a1, bu, au1);
    }
    async_wait();
    __syncthreads();
  }

  int n = n0 + wave * 16 + (lane & 15);
  int mbase = m0 + ((lane >> 4) * 8);
#pragma unroll
  for (int v = 0; v < 8; ++v) {
    float g0 = ag0[v], u0 = au0[v];
    float s0 = g0 * fast_rcp(1.f + __expf(-g0)) * u0;
    Out[(size_t)(mbase + v) * N + n] = f2bf(s0);
    float g1 = ag1[v], u1 = au1[v];
    float s1 = g1 * fast_rcp(1.f + __expf(-g1)) * u1;
    Out[(size_t)(mbase + 16 + v) * N + n] = f2bf(s1);
  }
}

// ---------------- fused attention per (b, h, 16-query tile) ----------------
// Q/K in [B,H,S,DH] bf16 (post-RoPE); V pre-transposed to [B,H,DH,S] bf16.
// O written token-major [B*S, H*DH] bf16.
template<int S>
__global__ __launch_bounds__(128) void attn_kernel(
    const bf16_t* __restrict__ Qh, const bf16_t* __restrict__ Kh,
    const bf16_t* __restrict__ Vt, bf16_t* __restrict__ O, int Hn)
{
  __shared__ float    sc[16][S];
  __shared__ uint32_t pb[16][S / 2];   // normalized P, bf16 pair-packed
  int q0 = blockIdx.x * 16;
  int h  = blockIdx.y;
  int b  = blockIdx.z;
  int tid = threadIdx.x, wave = tid >> 5, lane = tid & 31;
  const bf16_t* Qp = Qh + ((size_t)(b * Hn + h) * S) * DHH;
  const bf16_t* Kp = Kh + ((size_t)(b * Hn + h) * S) * DHH;
  const bf16_t* Vp = Vt + ((size_t)(b * Hn + h) * DHH) * S;

  // Q fragments for dh-steps [0,32) and [32,64): two b128 loads each
  v16bf qa0, qa1;
  {
    int m  = lane & 15;
    int kb = (lane >> 4) << 3;
    const bf16_t* row = Qp + (size_t)(q0 + m) * DHH;
    qa0 = make_frag(*(const uint4*)(row + kb),      *(const uint4*)(row + kb + 16));
    qa1 = make_frag(*(const uint4*)(row + 32 + kb), *(const uint4*)(row + 32 + kb + 16));
  }
  const float scale = 0.125f; // 1/sqrt(64)

  // scores: each wave covers key tiles kt, kt+64, ...
  for (int kt = wave * 16; kt < S; kt += 64) {
    v16bf bf0, bf1;
    {
      int n  = lane & 15;
      int kb = (lane >> 4) << 3;
      const bf16_t* col = Kp + (size_t)(kt + n) * DHH + kb;   // B[k=dh][n=key] = K[key][dh]
      bf0 = make_frag(*(const uint4*)(col),      *(const uint4*)(col + 16));
      bf1 = make_frag(*(const uint4*)(col + 32), *(const uint4*)(col + 48));
    }
    v8f acc = {};
    acc = wmma_bf16(qa0, bf0, acc);
    acc = wmma_bf16(qa1, bf1, acc);
    int n  = lane & 15;
    int mb = (lane >> 4) * 8;
#pragma unroll
    for (int v = 0; v < 8; ++v) {
      int qg = q0 + mb + v;
      int kg = kt + n;
      float s = acc[v] * scale;
      if (kg > qg) s = -1e9f;
      sc[mb + v][kg] = s;
    }
  }
  __syncthreads();

  // softmax: 8 consecutive threads per row; pack normalized P as bf16 pairs
  {
    int row = tid >> 3, j = tid & 7;
    float mx = -1e30f;
    for (int c = j; c < S; c += 8) mx = fmaxf(mx, sc[row][c]);
#pragma unroll
    for (int o = 4; o >= 1; o >>= 1) mx = fmaxf(mx, __shfl_xor(mx, o, 8));
    float sum = 0.f;
    for (int c = j; c < S; c += 8) { float e = __expf(sc[row][c] - mx); sc[row][c] = e; sum += e; }
#pragma unroll
    for (int o = 4; o >= 1; o >>= 1) sum += __shfl_xor(sum, o, 8);
    float inv = fast_rcp(sum);
    for (int c2 = j; c2 < S / 2; c2 += 8)
      pb[row][c2] = pack2(sc[row][2 * c2] * inv, sc[row][2 * c2 + 1] * inv);
  }
  __syncthreads();

  // O = P x V ; wave owns dh slice [wave*16, wave*16+16)
  v8f oacc = {};
  {
    int m   = lane & 15;
    int kp0 = (lane >> 4) << 2;            // pair base within 16-pair kstep
    int nl  = lane & 15;
    int kb  = (lane >> 4) << 3;
    const bf16_t* vrow = Vp + (size_t)(wave * 16 + nl) * S + kb;
    for (int ks2 = 0; ks2 < S / 2; ks2 += 16) {   // 32 keys per step
      const uint4* pp = (const uint4*)(&pb[m][ks2 + kp0]);
      v16bf pa = make_frag(pp[0], pp[2]);         // pairs +0..3 and +8..11
      const bf16_t* vc = vrow + 2 * ks2;
      v16bf vb = make_frag(*(const uint4*)(vc), *(const uint4*)(vc + 16));
      oacc = wmma_bf16(pa, vb, oacc);
    }
  }
  {
    int nl = lane & 15;
    int mb = (lane >> 4) * 8;
#pragma unroll
    for (int v = 0; v < 8; ++v) {
      int tok = b * S + q0 + mb + v;
      O[(size_t)tok * (Hn * DHH) + h * DHH + wave * 16 + nl] = f2bf(oacc[v]);
    }
  }
}

// ---------------- elementwise / reduction kernels ----------------
__global__ __launch_bounds__(256) void rmsnorm_kernel(
    const float* __restrict__ X, const float* __restrict__ W, bf16_t* __restrict__ Y, int D) {
  int row = blockIdx.x;
  const float* x = X + (size_t)row * D;
  bf16_t* y = Y + (size_t)row * D;
  __shared__ float red[256];
  float s = 0.f;
  for (int i = threadIdx.x; i < D; i += 256) { float v = x[i]; s += v * v; }
  red[threadIdx.x] = s; __syncthreads();
  for (int o = 128; o > 0; o >>= 1) { if (threadIdx.x < o) red[threadIdx.x] += red[threadIdx.x + o]; __syncthreads(); }
  float r = fast_rsq(red[0] / D + 1e-6f);
  for (int i = threadIdx.x * 2; i < D; i += 512) {
    float a = W[i] * x[i] * r;
    float b = W[i + 1] * x[i + 1] * r;
    *(uint32_t*)&y[i] = pack2(a, b);
  }
}

__global__ void pos_fill_kernel(float* pos, int T, int n) {
  int i = blockIdx.x * blockDim.x + threadIdx.x;
  if (i < n) pos[i] = (float)(i % T);
}

// RoPE + pack. Q/K -> [B,H,S,DH] bf16. V -> transposed [B,H,DH,S] bf16.
__global__ __launch_bounds__(256) void rope_pack_kernel(
    const float* __restrict__ Qf, const float* __restrict__ Kf, const float* __restrict__ Vf,
    const float* __restrict__ pos, bf16_t* __restrict__ Qh, bf16_t* __restrict__ Kh,
    bf16_t* __restrict__ Vt, int S, int Hn)
{
  int gt = blockIdx.x * 256 + threadIdx.x;     // B*S*Hn*32 threads
  int i  = gt & 31;
  int h  = (gt >> 5) % Hn;
  int tok = gt / (32 * Hn);
  int b = tok / S, s = tok - b * S;
  float p = pos[tok];
  float inv = __powf(10000.f, -(float)(2 * i) / 64.f);
  float a = p * inv;
  float c = __cosf(a), sn = __sinf(a);
  size_t src = (size_t)tok * (Hn * DHH) + h * DHH;
  size_t dst = (((size_t)b * Hn + h) * S + s) * DHH;
  float q1 = Qf[src + i], q2 = Qf[src + 32 + i];
  Qh[dst + i]      = f2bf(q1 * c - q2 * sn);
  Qh[dst + 32 + i] = f2bf(q2 * c + q1 * sn);
  float k1 = Kf[src + i], k2 = Kf[src + 32 + i];
  Kh[dst + i]      = f2bf(k1 * c - k2 * sn);
  Kh[dst + 32 + i] = f2bf(k2 * c + k1 * sn);
  size_t vb = ((size_t)(b * Hn + h) * DHH) * S;
  Vt[vb + (size_t)i * S + s]        = f2bf(Vf[src + i]);
  Vt[vb + (size_t)(i + 32) * S + s] = f2bf(Vf[src + 32 + i]);
}

// g_cont + per-token D_ch sum (prior_loss partials; (prior_hidden-processed)^2 == (a-p)^2)
__global__ __launch_bounds__(256) void router_kernel(
    const float* __restrict__ X, const float* __restrict__ P, const float* __restrict__ PR,
    float* __restrict__ g_cont, float* __restrict__ dch_tok, int D)
{
  int tok = blockIdx.x;
  const float* x  = X  + (size_t)tok * D;
  const float* pc = P  + (size_t)tok * D;
  const float* pr = PR + (size_t)tok * D;
  __shared__ float r1[256], r2[256];
  float sst = 0.f, sch = 0.f;
  for (int i = threadIdx.x; i < D; i += 256) {
    float a = pc[i] - x[i];
    float d = a - pr[i];
    sst += a * a; sch += d * d;
  }
  r1[threadIdx.x] = sst; r2[threadIdx.x] = sch; __syncthreads();
  for (int o = 128; o > 0; o >>= 1) {
    if (threadIdx.x < o) { r1[threadIdx.x] += r1[threadIdx.x + o]; r2[threadIdx.x] += r2[threadIdx.x + o]; }
    __syncthreads();
  }
  if (threadIdx.x == 0) {
    float Dst = r1[0] / D, Dch = r2[0] / D;
    g_cont[tok]  = fast_rcp(1.f + __expf(-(Dst - Dch)));
    dch_tok[tok] = r2[0];
  }
}

// top-K per batch row via rank count (reproduces jax top_k tie-breaking); indices come out sorted
__global__ __launch_bounds__(256) void topk_kernel(
    const float* __restrict__ g, int* __restrict__ flags,
    int* __restrict__ idx_s, float* __restrict__ gate_s, int T, int Kk)
{
  int b = blockIdx.x;
  const float* gb = g + (size_t)b * T;
  __shared__ float gs[TT];
  __shared__ int   fl[TT];
  for (int i = threadIdx.x; i < T; i += 256) gs[i] = gb[i];
  __syncthreads();
  for (int i = threadIdx.x; i < T; i += 256) {
    float gi = gs[i]; int cnt = 0;
    for (int j = 0; j < T; ++j) {
      float gj = gs[j];
      cnt += (gj > gi) || (gj == gi && j < i);
    }
    int sel = cnt < Kk;
    fl[i] = sel;
    flags[(size_t)b * T + i] = sel;
  }
  __syncthreads();
  for (int i = threadIdx.x; i < T; i += 256) {
    if (fl[i]) {
      int r = 0;
      for (int j = 0; j < i; ++j) r += fl[j];
      idx_s[(size_t)b * Kk + r]  = i;
      gate_s[(size_t)b * Kk + r] = gs[i];
    }
  }
}

__global__ __launch_bounds__(256) void closs_kernel(
    const float* __restrict__ X, const float* __restrict__ crw, const float* __restrict__ crb,
    const int* __restrict__ flags, float* __restrict__ bce_tok, int D)
{
  int tok = blockIdx.x;
  const float* x = X + (size_t)tok * D;
  __shared__ float red[256];
  float s = 0.f;
  for (int i = threadIdx.x; i < D; i += 256) s += x[i] * crw[i];
  red[threadIdx.x] = s; __syncthreads();
  for (int o = 128; o > 0; o >>= 1) { if (threadIdx.x < o) red[threadIdx.x] += red[threadIdx.x + o]; __syncthreads(); }
  if (threadIdx.x == 0) {
    float l = red[0] + crb[0];
    float t = (float)flags[tok];
    bce_tok[tok] = fmaxf(l, 0.f) - l * t + log1pf(expf(-fabsf(l)));
  }
}

__global__ __launch_bounds__(256) void reduce_mean_kernel(
    const float* __restrict__ v, int n, float scale, float* __restrict__ out)
{
  __shared__ float red[256];
  float s = 0.f;
  for (int i = threadIdx.x; i < n; i += 256) s += v[i];
  red[threadIdx.x] = s; __syncthreads();
  for (int o = 128; o > 0; o >>= 1) { if (threadIdx.x < o) red[threadIdx.x] += red[threadIdx.x + o]; __syncthreads(); }
  if (threadIdx.x == 0) out[0] = red[0] * scale;
}

__global__ __launch_bounds__(256) void gather_kernel(
    const float* __restrict__ Proc, const int* __restrict__ idx_s,
    float* __restrict__ sel, float* __restrict__ pos2, int T, int Kk, int D)
{
  int row = blockIdx.x;            // 0..B*K-1
  int b = row / Kk;
  int idx = idx_s[row];
  const float* src = Proc + ((size_t)(b * T + idx)) * D;
  float* dst = sel + (size_t)row * D;
  for (int i = threadIdx.x; i < D; i += 256) dst[i] = src[i];
  if (threadIdx.x == 0) pos2[row] = (float)idx;
}

__global__ void copy_kernel(const float* __restrict__ s, float* __restrict__ d, size_t n) {
  size_t i = (size_t)blockIdx.x * blockDim.x + threadIdx.x;
  size_t st = (size_t)gridDim.x * blockDim.x;
  for (; i < n; i += st) d[i] = s[i];
}

__global__ __launch_bounds__(256) void scatter_gated_kernel(
    const float* __restrict__ sel, const float* __restrict__ out2,
    const float* __restrict__ gate_s, const int* __restrict__ idx_s,
    float* __restrict__ fin, int T, int Kk, int D)
{
  int row = blockIdx.x;
  int b = row / Kk;
  int idx = idx_s[row];
  float g = gate_s[row];
  const float* s = sel + (size_t)row * D;
  const float* o = out2 + (size_t)row * D;
  float* d = fin + ((size_t)(b * T + idx)) * D;
  for (int i = threadIdx.x; i < D; i += 256) d[i] = s[i] + g * (o[i] - s[i]);
}

// ---------------- launch ----------------
extern "C" void kernel_launch(void* const* d_in, const int* in_sizes, int n_in,
                              void* d_out, int out_size, void* d_ws, size_t ws_size,
                              hipStream_t stream)
{
  (void)in_sizes; (void)n_in; (void)out_size; (void)ws_size;
  const float* X    = (const float*)d_in[0];
  const float* ln1  = (const float*)d_in[1];
  const float* wq1  = (const float*)d_in[2];
  const float* wk1  = (const float*)d_in[3];
  const float* wv1  = (const float*)d_in[4];
  const float* wo1  = (const float*)d_in[5];
  const float* ln2  = (const float*)d_in[6];
  const float* g1   = (const float*)d_in[7];
  const float* u1   = (const float*)d_in[8];
  const float* dn1  = (const float*)d_in[9];
  const float* pln  = (const float*)d_in[10];
  const float* pg   = (const float*)d_in[11];
  const float* pu   = (const float*)d_in[12];
  const float* pd   = (const float*)d_in[13];
  const float* ln3  = (const float*)d_in[14];
  const float* wq2  = (const float*)d_in[15];
  const float* wk2  = (const float*)d_in[16];
  const float* wv2  = (const float*)d_in[17];
  const float* wo2  = (const float*)d_in[18];
  const float* ln4  = (const float*)d_in[19];
  const float* g2   = (const float*)d_in[20];
  const float* u2   = (const float*)d_in[21];
  const float* dn2  = (const float*)d_in[22];
  const float* crw  = (const float*)d_in[23];
  const float* crb  = (const float*)d_in[24];
  float* out = (float*)d_out;

  const int T1 = BB * TT;     // 4096 tokens, layer 1
  const int T2 = BB * KSEL;   // 1024 tokens, routed layer
  const size_t DDn = (size_t)DMODEL * DMODEL;
  const size_t DFn = (size_t)DMODEL * FF;

  char* base = (char*)d_ws;
  size_t off = 0;
  auto alloc = [&](size_t bytes) -> void* {
    void* p = base + off;
    off = (off + bytes + 255) & ~(size_t)255;
    return p;
  };

  // packed bf16 weight copies (pair-packed tile layout, uint32 per k-pair)
  uint32_t* wq1b = (uint32_t*)alloc(DDn * 2);
  uint32_t* wk1b = (uint32_t*)alloc(DDn * 2);
  uint32_t* wv1b = (uint32_t*)alloc(DDn * 2);
  uint32_t* wo1b = (uint32_t*)alloc(DDn * 2);
  uint32_t* g1b  = (uint32_t*)alloc(DFn * 2);
  uint32_t* u1b  = (uint32_t*)alloc(DFn * 2);
  uint32_t* d1b  = (uint32_t*)alloc(DFn * 2);
  uint32_t* pgb  = (uint32_t*)alloc(DDn * 2);
  uint32_t* pub  = (uint32_t*)alloc(DDn * 2);
  uint32_t* pdb  = (uint32_t*)alloc(DDn * 2);
  uint32_t* wq2b = (uint32_t*)alloc(DDn * 2);
  uint32_t* wk2b = (uint32_t*)alloc(DDn * 2);
  uint32_t* wv2b = (uint32_t*)alloc(DDn * 2);
  uint32_t* wo2b = (uint32_t*)alloc(DDn * 2);
  uint32_t* g2b  = (uint32_t*)alloc(DFn * 2);
  uint32_t* u2b  = (uint32_t*)alloc(DFn * 2);
  uint32_t* d2b  = (uint32_t*)alloc(DFn * 2);

  // activations (layer-2 reuses the big ones)
  const size_t A1 = (size_t)T1 * DMODEL;
  bf16_t* xn_bf   = (bf16_t*)alloc(A1 * 2);
  float*  q_f     = (float*)alloc(A1 * 4);
  float*  k_f     = (float*)alloc(A1 * 4);
  float*  v_f     = (float*)alloc(A1 * 4);
  bf16_t* qh      = (bf16_t*)alloc(A1 * 2);
  bf16_t* kh      = (bf16_t*)alloc(A1 * 2);
  bf16_t* vt      = (bf16_t*)alloc(A1 * 2);
  bf16_t* o_bf    = (bf16_t*)alloc(A1 * 2);
  float*  h_f     = (float*)alloc(A1 * 4);
  bf16_t* hn_bf   = (bf16_t*)alloc(A1 * 2);
  bf16_t* act_bf  = (bf16_t*)alloc((size_t)T1 * FF * 2);
  float*  proc    = (float*)alloc(A1 * 4);
  bf16_t* pn_bf   = (bf16_t*)alloc(A1 * 2);
  bf16_t* pact_bf = (bf16_t*)alloc((size_t)T1 * FPP * 2);
  float*  prior_f = (float*)alloc(A1 * 4);
  float*  g_cont  = (float*)alloc((size_t)T1 * 4);
  float*  dch_tok = (float*)alloc((size_t)T1 * 4);
  float*  bce_tok = (float*)alloc((size_t)T1 * 4);
  int*    flags   = (int*)alloc((size_t)T1 * 4);
  int*    idx_s   = (int*)alloc((size_t)T2 * 4);
  float*  gate_s  = (float*)alloc((size_t)T2 * 4);
  float*  pos1    = (float*)alloc((size_t)T1 * 4);
  float*  pos2    = (float*)alloc((size_t)T2 * 4);
  float*  sel_f   = (float*)alloc((size_t)T2 * DMODEL * 4);
  float*  out2_f  = (float*)alloc((size_t)T2 * DMODEL * 4);

  auto pack = [&](const float* w, uint32_t* wp, int nshift, size_t kdim) {
    size_t total = (kdim << nshift) / 2;     // K*N/2 dwords
    pack_weight_kernel<<<4096, 256, 0, stream>>>(w, wp, nshift, total);
  };
  auto gemm = [&](const bf16_t* A, const uint32_t* Bm, const float* res, float* C,
                  int M, int N, int Kd) {
    if (res)
      gemm_bf16_kernel<true><<<dim3(M / 32, N / 256), 256, 0, stream>>>(A, Bm, res, C, M, N, Kd);
    else
      gemm_bf16_kernel<false><<<dim3(M / 32, N / 256), 256, 0, stream>>>(A, Bm, nullptr, C, M, N, Kd);
  };
  auto swiglu = [&](const bf16_t* A, const uint32_t* Bg, const uint32_t* Bu, bf16_t* O,
                    int M, int N, int Kd) {
    gemm_swiglu_kernel<<<dim3(M / 32, N / 128), 256, 0, stream>>>(A, Bg, Bu, O, M, N, Kd);
  };

  // weight packing (N = 1<<nshift)
  pack(wq1, wq1b, 10, DMODEL); pack(wk1, wk1b, 10, DMODEL);
  pack(wv1, wv1b, 10, DMODEL); pack(wo1, wo1b, 10, DMODEL);
  pack(g1, g1b, 12, DMODEL);   pack(u1, u1b, 12, DMODEL);   pack(dn1, d1b, 10, FF);
  pack(pg, pgb, 10, DMODEL);   pack(pu, pub, 10, DMODEL);   pack(pd, pdb, 10, FPP);
  pack(wq2, wq2b, 10, DMODEL); pack(wk2, wk2b, 10, DMODEL);
  pack(wv2, wv2b, 10, DMODEL); pack(wo2, wo2b, 10, DMODEL);
  pack(g2, g2b, 12, DMODEL);   pack(u2, u2b, 12, DMODEL);   pack(dn2, d2b, 10, FF);

  pos_fill_kernel<<<(T1 + 255) / 256, 256, 0, stream>>>(pos1, TT, T1);

  // ---- decoder layer 1 ----
  rmsnorm_kernel<<<T1, 256, 0, stream>>>(X, ln1, xn_bf, DMODEL);
  gemm(xn_bf, wq1b, nullptr, q_f, T1, DMODEL, DMODEL);
  gemm(xn_bf, wk1b, nullptr, k_f, T1, DMODEL, DMODEL);
  gemm(xn_bf, wv1b, nullptr, v_f, T1, DMODEL, DMODEL);
  rope_pack_kernel<<<T1 * 2, 256, 0, stream>>>(q_f, k_f, v_f, pos1, qh, kh, vt, TT, HH);
  attn_kernel<TT><<<dim3(TT / 16, HH, BB), 128, 0, stream>>>(qh, kh, vt, o_bf, HH);
  gemm(o_bf, wo1b, X, h_f, T1, DMODEL, DMODEL);                 // h = x + attn
  rmsnorm_kernel<<<T1, 256, 0, stream>>>(h_f, ln2, hn_bf, DMODEL);
  swiglu(hn_bf, g1b, u1b, act_bf, T1, FF, DMODEL);
  gemm(act_bf, d1b, h_f, proc, T1, DMODEL, FF);                 // processed

  // ---- prior network ----
  rmsnorm_kernel<<<T1, 256, 0, stream>>>(X, pln, pn_bf, DMODEL);
  swiglu(pn_bf, pgb, pub, pact_bf, T1, FPP, DMODEL);
  gemm(pact_bf, pdb, nullptr, prior_f, T1, DMODEL, FPP);        // prior_out

  // ---- router, top-k, losses ----
  router_kernel<<<T1, 256, 0, stream>>>(X, proc, prior_f, g_cont, dch_tok, DMODEL);
  reduce_mean_kernel<<<1, 256, 0, stream>>>(dch_tok, T1, 1.f / ((float)T1 * DMODEL),
                                            out + (size_t)T1 * DMODEL);
  topk_kernel<<<BB, 256, 0, stream>>>(g_cont, flags, idx_s, gate_s, TT, KSEL);
  closs_kernel<<<T1, 256, 0, stream>>>(X, crw, crb, flags, bce_tok, DMODEL);
  reduce_mean_kernel<<<1, 256, 0, stream>>>(bce_tok, T1, 1.f / (float)T1,
                                            out + (size_t)T1 * DMODEL + 1);

  // ---- gather selected tokens ----
  gather_kernel<<<T2, 256, 0, stream>>>(proc, idx_s, sel_f, pos2, TT, KSEL, DMODEL);

  // ---- decoder layer 2 (reuses layer-1 scratch) ----
  rmsnorm_kernel<<<T2, 256, 0, stream>>>(sel_f, ln3, xn_bf, DMODEL);
  gemm(xn_bf, wq2b, nullptr, q_f, T2, DMODEL, DMODEL);
  gemm(xn_bf, wk2b, nullptr, k_f, T2, DMODEL, DMODEL);
  gemm(xn_bf, wv2b, nullptr, v_f, T2, DMODEL, DMODEL);
  rope_pack_kernel<<<T2 * 2, 256, 0, stream>>>(q_f, k_f, v_f, pos2, qh, kh, vt, KSEL, HH);
  attn_kernel<KSEL><<<dim3(KSEL / 16, HH, BB), 128, 0, stream>>>(qh, kh, vt, o_bf, HH);
  gemm(o_bf, wo2b, sel_f, h_f, T2, DMODEL, DMODEL);             // h2 = sel + attn
  rmsnorm_kernel<<<T2, 256, 0, stream>>>(h_f, ln4, hn_bf, DMODEL);
  swiglu(hn_bf, g2b, u2b, act_bf, T2, FF, DMODEL);
  gemm(act_bf, d2b, h_f, out2_f, T2, DMODEL, FF);               // out2

  // ---- final: copy processed, then soft-gated scatter of routed tokens ----
  copy_kernel<<<4096, 256, 0, stream>>>(proc, out, (size_t)T1 * DMODEL);
  scatter_gated_kernel<<<T2, 256, 0, stream>>>(sel_f, out2_f, gate_s, idx_s, out,
                                               TT, KSEL, DMODEL);
}
